// MoELoRALinearLayer_61856118997284
// MI455X (gfx1250) — matrix operand order repaired and checked by reading, not compiled, not run on hardware
//
#include <hip/hip_runtime.h>

// ---- problem constants (match reference) ----
#define DD 1024
#define EE 64
#define RR 8
#define TT 16384
#define SCALING (1.0f / 8.0f)

typedef __attribute__((ext_vector_type(16))) __bf16 v16bf;
typedef __attribute__((ext_vector_type(4)))  __bf16 v4bf;
typedef __attribute__((ext_vector_type(8)))  float  v8f;
typedef __attribute__((ext_vector_type(4)))  float  v4f;

union AF { v16bf v; v4bf q[4]; };
union BF { v16bf v; uint4 u[2]; };

// ---------------------------------------------------------------------------
// Kernel 1: xa[t, r] = SCALING * sum_k x[t,k] * A[labels[t], k, r]
// one wave (32 lanes) per token, cross-lane shuffle reduction
// ---------------------------------------------------------------------------
__global__ __launch_bounds__(256)
void moelora_xa_kernel(const float* __restrict__ x, const int* __restrict__ labels,
                       const float* __restrict__ A, float* __restrict__ xa) {
  const int lane = threadIdx.x & 31;
  const int wave = threadIdx.x >> 5;
  const int t = blockIdx.x * 8 + wave;
  const int l = labels[t];
  const float* __restrict__ xrow = x + (size_t)t * DD;
  const float* __restrict__ Ae = A + (size_t)l * DD * RR;
  float acc[RR];
#pragma unroll
  for (int r = 0; r < RR; ++r) acc[r] = 0.0f;
  for (int k = lane; k < DD; k += 32) {
    const float xv = xrow[k];
    const float* __restrict__ Ar = Ae + k * RR;
#pragma unroll
    for (int r = 0; r < RR; ++r) acc[r] += xv * Ar[r];
  }
#pragma unroll
  for (int r = 0; r < RR; ++r) {
#pragma unroll
    for (int off = 16; off > 0; off >>= 1)
      acc[r] += __shfl_xor(acc[r], off, 32);
  }
  if (lane == 0) {
#pragma unroll
    for (int r = 0; r < RR; ++r) xa[t * RR + r] = SCALING * acc[r];
  }
}

// ---------------------------------------------------------------------------
// Kernel 2: WbT[n*D + k] = bf16(W[k*D + n])  (transpose + convert, 2 MB out)
// coalesced writes; W is small and L2-resident so strided reads are cheap
// ---------------------------------------------------------------------------
__global__ __launch_bounds__(256)
void moelora_wt_kernel(const float* __restrict__ W, __bf16* __restrict__ WbT) {
  const int idx = blockIdx.x * 256 + threadIdx.x;   // idx = n*D + k
  const int n = idx >> 10;
  const int k = idx & (DD - 1);
  WbT[idx] = (__bf16)W[k * DD + n];
}

// ---------------------------------------------------------------------------
// Kernel 3: y = bf16GEMM(x, W) + xa @ B[labels] + bias
// 128x128 output tile per block; 8 waves; wave = 2x4 grid of 16x16x32 WMMA
// ---------------------------------------------------------------------------
__global__ __launch_bounds__(256)
void moelora_gemm_kernel(const float* __restrict__ x,
                         const __bf16* __restrict__ WbT,
                         const int* __restrict__ labels,
                         const float* __restrict__ B,
                         const float* __restrict__ bias,
                         const float* __restrict__ xa,
                         float* __restrict__ out) {
  const int lane  = threadIdx.x & 31;
  const int wave  = threadIdx.x >> 5;
  const int waveM = wave & 3;   // 4 waves along M  -> 4*32 = 128 rows
  const int waveN = wave >> 2;  // 2 waves along N  -> 2*64 = 128 cols
  const int lm    = lane & 15;
  const int kh    = lane >> 4;  // K-half selector per WMMA layout

  const int rowBase = blockIdx.y * 128 + waveM * 32;
  const int colBase = blockIdx.x * 128 + waveN * 64;

  v8f acc[2][4];
#pragma unroll
  for (int tm = 0; tm < 2; ++tm)
#pragma unroll
    for (int tn = 0; tn < 4; ++tn)
      acc[tm][tn] = (v8f){0.f, 0.f, 0.f, 0.f, 0.f, 0.f, 0.f, 0.f};

  for (int kt = 0; kt < DD; kt += 32) {
    // --- A fragments (x rows, fp32 -> bf16 via hw packed converts) ---
    // lane lm (+16*kh): row = base+lm, holds K = 16*kh + 0..15
    AF af[2];
#pragma unroll
    for (int tm = 0; tm < 2; ++tm) {
      const int row = rowBase + tm * 16 + lm;
      const float* __restrict__ p = x + (size_t)row * DD + kt + kh * 16;
      const v4f f0 = *(const v4f*)(p);
      const v4f f1 = *(const v4f*)(p + 4);
      const v4f f2 = *(const v4f*)(p + 8);
      const v4f f3 = *(const v4f*)(p + 12);
      af[tm].q[0] = __builtin_convertvector(f0, v4bf);
      af[tm].q[1] = __builtin_convertvector(f1, v4bf);
      af[tm].q[2] = __builtin_convertvector(f2, v4bf);
      af[tm].q[3] = __builtin_convertvector(f3, v4bf);
    }
    // --- B fragments from pre-transposed bf16 W: contiguous 32B per lane ---
    // lane lm (+16*kh): column = base+lm, holds K = 16*kh + 0..15
    BF bf[4];
#pragma unroll
    for (int tn = 0; tn < 4; ++tn) {
      const int col = colBase + tn * 16 + lm;
      const uint4* __restrict__ q =
          (const uint4*)(WbT + (size_t)col * DD + kt + kh * 16);
      bf[tn].u[0] = q[0];
      bf[tn].u[1] = q[1];
    }
    // --- 8 WMMAs: D = A x B + C (fp32 accumulate) ---
#pragma unroll
    for (int tm = 0; tm < 2; ++tm)
#pragma unroll
      for (int tn = 0; tn < 4; ++tn)
        acc[tm][tn] = __builtin_amdgcn_wmma_f32_16x16x32_bf16(
            /*neg_a=*/false, af[tm].v, /*neg_b=*/false, bf[tn].v,
            /*c_mod=*/(short)0, acc[tm][tn],
            /*reuse_a=*/false, /*reuse_b=*/false);
  }

  // --- epilogue: + SCALING*(x.A[l]) @ B[l]  (xa already scaled) + bias ---
  // C layout: row = tile + 8*(lane/16) + j, col = tile + lane%16
  float bv[4];
#pragma unroll
  for (int tn = 0; tn < 4; ++tn) bv[tn] = bias[colBase + tn * 16 + lm];

#pragma unroll
  for (int tm = 0; tm < 2; ++tm) {
#pragma unroll
    for (int j = 0; j < 8; ++j) {
      const int row = rowBase + tm * 16 + kh * 8 + j;
      const int l = labels[row];                    // loaded once per row
      const float* __restrict__ xap = xa + row * RR;
      float xv[RR];
#pragma unroll
      for (int r = 0; r < RR; ++r) xv[r] = xap[r];  // loaded once per row
      const float* __restrict__ Bl = B + (size_t)l * RR * DD;
      float* __restrict__ orow = out + (size_t)row * DD;
#pragma unroll
      for (int tn = 0; tn < 4; ++tn) {
        const int col = colBase + tn * 16 + lm;
        float c = acc[tm][tn][j] + bv[tn];
#pragma unroll
        for (int r = 0; r < RR; ++r) c += xv[r] * Bl[r * DD + col];
        orow[col] = c;
      }
    }
  }
}

// ---------------------------------------------------------------------------
extern "C" void kernel_launch(void* const* d_in, const int* in_sizes, int n_in,
                              void* d_out, int out_size, void* d_ws, size_t ws_size,
                              hipStream_t stream) {
  const float* x      = (const float*)d_in[0];
  const int*   labels = (const int*)d_in[1];
  const float* W      = (const float*)d_in[2];
  const float* A      = (const float*)d_in[3];
  const float* B      = (const float*)d_in[4];
  const float* bias   = (const float*)d_in[5];
  float* out = (float*)d_out;

  // workspace: [0, 2MB) bf16 W^T ; [2MB, 2.5MB) xa[T][R] fp32
  __bf16* WbT = (__bf16*)d_ws;
  float* xa = (float*)((char*)d_ws + (size_t)DD * DD * sizeof(__bf16));

  moelora_wt_kernel<<<(DD * DD) / 256, 256, 0, stream>>>(W, WbT);
  moelora_xa_kernel<<<TT / 8, 256, 0, stream>>>(x, labels, A, xa);

  dim3 grid(DD / 128, TT / 128);
  moelora_gemm_kernel<<<grid, 256, 0, stream>>>(x, WbT, labels, B, bias, xa, out);
}
